// BlockwiseEarlyExitMamba_26319559590234
// MI455X (gfx1250) — compile-verified, matrix operand order checked by reading, not run on hardware
//
#include <hip/hip_runtime.h>
#include <hip/hip_bf16.h>
#include <math.h>

// ---------------- model constants ----------------
#define D_MODEL   256
#define D_INNER   512
#define D_STATE   16
#define D_CONV    4
#define DT_RANK   16
#define N_LAYERS  4
#define BATCH     2048
#define SEQ       32
#define NTOK      (BATCH * SEQ)          // 65536
#define D_CAT     136
#define D_CATP    160                    // padded to multiple of 32

typedef _Float16 half_t;
typedef __attribute__((ext_vector_type(16))) _Float16 v16h;
typedef __attribute__((ext_vector_type(8)))  _Float16 v8h;
typedef __attribute__((ext_vector_type(8)))  float    v8f;

__device__ __forceinline__ float siluf(float v) {
    return v * (1.0f / (1.0f + __expf(-v)));
}

// gfx1250 async global->LDS, 16B per enabled lane
__device__ __forceinline__ void async_load_b128(unsigned lds_off, const void* gaddr) {
    asm volatile("global_load_async_to_lds_b128 %0, %1, off"
                 :: "v"(lds_off), "v"(gaddr) : "memory");
}
__device__ __forceinline__ void wait_asynccnt0() {
    asm volatile("s_wait_asynccnt 0x0" ::: "memory");
}

// =================================================================
// 0) generic f32 -> f16 weight conversion with K padding
//    src: rows x Ks (f32), dst: rows x Kd (f16, zero padded)
// =================================================================
__global__ void cvt_pad_kernel(const float* __restrict__ src, int Ks,
                               _Float16* __restrict__ dst, int Kd, int rows) {
    int i = blockIdx.x * blockDim.x + threadIdx.x;
    if (i >= rows * Kd) return;
    int r = i / Kd;
    int c = i - r * Kd;
    dst[i] = (c < Ks) ? (_Float16)src[(size_t)r * Ks + c] : (_Float16)0.0f;
}

// =================================================================
// 1) Build the padded 160-wide fused input feature
// =================================================================
__global__ void build_cat_kernel(const float* __restrict__ x,
                                 const float* __restrict__ emb_proto,
                                 const float* __restrict__ emb_flags,
                                 const float* __restrict__ emb_dir,
                                 const float* __restrict__ proj_len_w,
                                 const float* __restrict__ proj_len_b,
                                 const float* __restrict__ proj_iat_w,
                                 const float* __restrict__ proj_iat_b,
                                 float* __restrict__ cat) {
    int i = blockIdx.x * blockDim.x + threadIdx.x;
    if (i >= NTOK * D_CATP) return;
    int t = i / D_CATP;
    int f = i - t * D_CATP;
    const float* xr = x + (size_t)t * 5;
    float val = 0.0f;
    if (f < 32) {
        int p = (int)xr[0]; p = p < 0 ? 0 : (p > 255 ? 255 : p);
        val = emb_proto[p * 32 + f];
    } else if (f < 64) {
        int j = f - 32;
        val = xr[1] * proj_len_w[j] + proj_len_b[j];
    } else if (f < 96) {
        int fl = (int)xr[2]; fl = fl < 0 ? 0 : (fl > 63 ? 63 : fl);
        val = emb_flags[fl * 32 + (f - 64)];
    } else if (f < 128) {
        int j = f - 96;
        val = xr[3] * proj_iat_w[j] + proj_iat_b[j];
    } else if (f < 136) {
        int d = (int)xr[4]; d = d < 0 ? 0 : (d > 1 ? 1 : d);
        val = emb_dir[d * 8 + (f - 128)];
    }
    cat[(size_t)t * D_CATP + f] = val;
}

// =================================================================
// 2) WMMA GEMM:  C[M,N] = act( A[M,K] @ Wh[N,K]^T + bias )
//    A f32 row-major (row stride lda, 16B-aligned rows, K % 32 == 0,
//    M % 128 == 0). Wh f16 row-major (N x K). C f32, row stride ldc.
//    Block 256 thr (8 waves). Tile 128(M) x 64(N), K-step 32.
//    A tile staged with vectorized cvt; B tile staged with
//    global_load_async_to_lds_b128 (+ s_wait_asynccnt).
// =================================================================
#define MT 128
#define NT 64
#define KT 32

__global__ __launch_bounds__(256)
void wmma_gemm_kernel(const float* __restrict__ A, int lda,
                      const _Float16* __restrict__ Wh,
                      const float* __restrict__ bias,
                      float* __restrict__ C, int ldc,
                      int M, int N, int K, int act) {
    __shared__ __align__(16) _Float16 As[MT * KT];
    __shared__ __align__(16) _Float16 Bs[NT * KT];

    const int tid    = threadIdx.x;
    const int lane   = tid & 31;
    const int wave   = tid >> 5;
    const int waveM  = wave & 3;
    const int waveN  = wave >> 2;
    const int lrow   = lane & 15;
    const int halfId = lane >> 4;
    const int kc     = halfId * 8;

    const int mBase = blockIdx.y * MT;
    const int nBase = blockIdx.x * NT;

    // A staging: thread -> one 16-wide chunk of one row
    const int arow = tid >> 1;
    const int acol = (tid & 1) * 16;
    // B staging: thread -> one 8-wide (16B) chunk of one row
    const int brow = tid >> 2;
    const int bcol = (tid & 3) * 8;
    const int gn   = nBase + brow;
    const unsigned bs_lds = (unsigned)(uintptr_t)(&Bs[brow * KT + bcol]);

    // zero-fill OOB B rows once (stay zero; async loads skip disabled lanes)
    if (nBase + NT > N) {
        for (int idx = tid; idx < NT * KT; idx += 256) {
            int r = idx >> 5;
            if (nBase + r >= N) Bs[idx] = (_Float16)0.0f;
        }
    }
    __syncthreads();

    v8f acc[2][2] = {};

    for (int k0 = 0; k0 < K; k0 += KT) {
        // ---- stage A tile (128x32): 4x float4 loads + 2x b128 LDS stores ----
        {
            const float* ag = A + (size_t)(mBase + arow) * lda + k0 + acol;
            float4 f0 = *(const float4*)(ag + 0);
            float4 f1 = *(const float4*)(ag + 4);
            float4 f2 = *(const float4*)(ag + 8);
            float4 f3 = *(const float4*)(ag + 12);
            v8h h0, h1;
            h0[0] = (_Float16)f0.x; h0[1] = (_Float16)f0.y;
            h0[2] = (_Float16)f0.z; h0[3] = (_Float16)f0.w;
            h0[4] = (_Float16)f1.x; h0[5] = (_Float16)f1.y;
            h0[6] = (_Float16)f1.z; h0[7] = (_Float16)f1.w;
            h1[0] = (_Float16)f2.x; h1[1] = (_Float16)f2.y;
            h1[2] = (_Float16)f2.z; h1[3] = (_Float16)f2.w;
            h1[4] = (_Float16)f3.x; h1[5] = (_Float16)f3.y;
            h1[6] = (_Float16)f3.z; h1[7] = (_Float16)f3.w;
            *(v8h*)&As[arow * KT + acol]     = h0;
            *(v8h*)&As[arow * KT + acol + 8] = h1;
        }
        // ---- stage B tile (64x32) via async global->LDS (f16 direct) ----
        if (gn < N) {
            async_load_b128(bs_lds, Wh + (size_t)gn * K + k0 + bcol);
        }
        wait_asynccnt0();
        __syncthreads();

        // ---- fragments per ISA 16-bit layout, accumulate ----
        v16h afr[2], bfr[2];
#pragma unroll
        for (int i = 0; i < 2; ++i) {
            const _Float16* p = &As[(waveM * 32 + i * 16 + lrow) * KT + kc];
            v8h lo = *(const v8h*)p;
            v8h hi = *(const v8h*)(p + 16);
            afr[i] = __builtin_shufflevector(lo, hi, 0,1,2,3,4,5,6,7,8,9,10,11,12,13,14,15);
        }
#pragma unroll
        for (int j = 0; j < 2; ++j) {
            const _Float16* p = &Bs[(waveN * 32 + j * 16 + lrow) * KT + kc];
            v8h lo = *(const v8h*)p;
            v8h hi = *(const v8h*)(p + 16);
            bfr[j] = __builtin_shufflevector(lo, hi, 0,1,2,3,4,5,6,7,8,9,10,11,12,13,14,15);
        }
#pragma unroll
        for (int i = 0; i < 2; ++i)
#pragma unroll
            for (int j = 0; j < 2; ++j)
                acc[i][j] = __builtin_amdgcn_wmma_f32_16x16x32_f16(
                    false, afr[i], false, bfr[j], (short)0, acc[i][j], false, false);
        __syncthreads();
    }

    // ---- epilogue: bias + activation + store ----
#pragma unroll
    for (int j = 0; j < 2; ++j) {
        int n = nBase + waveN * 32 + j * 16 + lrow;
        if (n >= N) continue;
        float bb = (bias != nullptr) ? bias[n] : 0.0f;
#pragma unroll
        for (int i = 0; i < 2; ++i) {
            int m0 = mBase + waveM * 32 + i * 16 + 8 * halfId;
#pragma unroll
            for (int r = 0; r < 8; ++r) {
                float v = acc[i][j][r] + bb;
                if (act == 1) v = (v > 20.0f) ? v : log1pf(__expf(v));
                C[(size_t)(m0 + r) * ldc + n] = v;
            }
        }
    }
}

// =================================================================
// 3) Residual add + LayerNorm over D_MODEL=256. One wave per token.
// =================================================================
__global__ __launch_bounds__(256)
void add_ln_kernel(const float* __restrict__ src,
                   const float* __restrict__ resid,   // may be null
                   const float* __restrict__ g,
                   const float* __restrict__ b,
                   float* __restrict__ out) {
    int wave = threadIdx.x >> 5;
    int lane = threadIdx.x & 31;
    int tok  = blockIdx.x * 8 + wave;
    if (tok >= NTOK) return;
    size_t base = (size_t)tok * D_MODEL;

    float v[8];
    float s = 0.0f;
#pragma unroll
    for (int r = 0; r < 8; ++r) {
        int e = lane + r * 32;
        float val = src[base + e];
        if (resid) val += resid[base + e];
        v[r] = val;
        s += val;
    }
#pragma unroll
    for (int off = 16; off > 0; off >>= 1) s += __shfl_xor(s, off, 32);
    float mean = s * (1.0f / D_MODEL);
    float s2 = 0.0f;
#pragma unroll
    for (int r = 0; r < 8; ++r) { float d = v[r] - mean; s2 += d * d; }
#pragma unroll
    for (int off = 16; off > 0; off >>= 1) s2 += __shfl_xor(s2, off, 32);
    float rs = rsqrtf(s2 * (1.0f / D_MODEL) + 1e-5f);
#pragma unroll
    for (int r = 0; r < 8; ++r) {
        int e = lane + r * 32;
        out[base + e] = (v[r] - mean) * rs * g[e] + b[e];
    }
}

// =================================================================
// 4) Depthwise causal conv (width 4) + bias + SiLU, in place on xi.
// =================================================================
__global__ void conv_silu_kernel(float* __restrict__ xz,
                                 const float* __restrict__ cw,
                                 const float* __restrict__ cb) {
    int idx = blockIdx.x * blockDim.x + threadIdx.x;
    if (idx >= BATCH * D_INNER) return;
    int b = idx >> 9;
    int d = idx & (D_INNER - 1);
    float c0 = cw[d * 4 + 0], c1 = cw[d * 4 + 1], c2 = cw[d * 4 + 2], c3 = cw[d * 4 + 3];
    float cbd = cb[d];
    float* base = xz + (size_t)b * SEQ * (2 * D_INNER) + d;
    float x0 = 0.f, x1 = 0.f, x2 = 0.f;
#pragma unroll
    for (int s = 0; s < SEQ; ++s) {
        float x3 = base[(size_t)s * (2 * D_INNER)];
        float v = x0 * c0 + x1 * c1 + x2 * c2 + x3 * c3 + cbd;
        base[(size_t)s * (2 * D_INNER)] = siluf(v);
        x0 = x1; x1 = x2; x2 = x3;
    }
}

// =================================================================
// 5) Repack dt-rank columns of dbl into zero-padded 32-wide buffer
// =================================================================
__global__ void repack_dt_kernel(const float* __restrict__ dbl,
                                 float* __restrict__ dt16) {
    int i = blockIdx.x * blockDim.x + threadIdx.x;
    if (i >= NTOK * 32) return;
    int row = i >> 5;
    int col = i & 31;
    dt16[i] = (col < DT_RANK) ? dbl[(size_t)row * (DT_RANK + 2 * D_STATE) + col] : 0.0f;
}

// =================================================================
// 6) Selective scan. One thread per (batch, channel), h[16] in regs.
// =================================================================
__global__ void scan_kernel(const float* __restrict__ xz,
                            const float* __restrict__ dbl,
                            float* __restrict__ dt_y,       // dt in, y out (aliased)
                            const float* __restrict__ A_log,
                            const float* __restrict__ Dp) {
    int idx = blockIdx.x * blockDim.x + threadIdx.x;
    if (idx >= BATCH * D_INNER) return;
    int b = idx >> 9;
    int d = idx & (D_INNER - 1);

    float Ad[D_STATE];
#pragma unroll
    for (int n = 0; n < D_STATE; ++n) Ad[n] = -__expf(A_log[d * D_STATE + n]);
    float h[D_STATE];
#pragma unroll
    for (int n = 0; n < D_STATE; ++n) h[n] = 0.0f;
    float Dd = Dp[d];

    for (int s = 0; s < SEQ; ++s) {
        size_t row = (size_t)b * SEQ + s;
        float dtv = dt_y[row * D_INNER + d];
        float xv  = xz[row * (2 * D_INNER) + d];
        float zv  = xz[row * (2 * D_INNER) + D_INNER + d];
        const float* dr = dbl + row * (DT_RANK + 2 * D_STATE);
        float dx = dtv * xv;
        float yv = 0.0f;
#pragma unroll
        for (int n = 0; n < D_STATE; ++n) {
            float Bn = dr[DT_RANK + n];
            float Cn = dr[DT_RANK + D_STATE + n];
            h[n] = h[n] * __expf(dtv * Ad[n]) + dx * Bn;
            yv += h[n] * Cn;
        }
        yv += Dd * xv;
        yv *= siluf(zv);
        dt_y[row * D_INNER + d] = yv;
    }
}

// =================================================================
// 7) Early-exit heads: one block (128 threads) per batch element.
// =================================================================
__global__ __launch_bounds__(128)
void heads_kernel(const float* __restrict__ feat,
                  const float* __restrict__ clf_w1, const float* __restrict__ clf_b1,
                  const float* __restrict__ clf_w2, const float* __restrict__ clf_b2,
                  const float* __restrict__ conf_w1, const float* __restrict__ conf_b1,
                  const float* __restrict__ conf_w2, const float* __restrict__ conf_b2,
                  float* __restrict__ out) {
    __shared__ float fr[3][D_MODEL];
    __shared__ float h1[128];
    __shared__ float lg[3][2];
    __shared__ float cf[2];

    const int b = blockIdx.x;
    const int tid = threadIdx.x;
    const int idxs[3] = {7, 15, 31};

    for (int i = 0; i < 3; ++i)
        for (int e = tid; e < D_MODEL; e += 128)
            fr[i][e] = feat[((size_t)b * SEQ + idxs[i]) * D_MODEL + e];
    __syncthreads();

    for (int i = 0; i < 3; ++i) {
        float a = clf_b1[i * 128 + tid];
        const float* w = clf_w1 + ((size_t)i * 128 + tid) * D_MODEL;
        for (int k = 0; k < D_MODEL; ++k) a += fr[i][k] * w[k];
        h1[tid] = fmaxf(a, 0.0f);
        __syncthreads();
        if (tid < 2) {
            float a2 = clf_b2[i * 2 + tid];
            const float* w2 = clf_w2 + ((size_t)i * 2 + tid) * 128;
            for (int k = 0; k < 128; ++k) a2 += h1[k] * w2[k];
            lg[i][tid] = a2;
        }
        __syncthreads();
    }

    for (int i = 0; i < 2; ++i) {
        if (tid < 64) {
            float a = conf_b1[i * 64 + tid];
            const float* w = conf_w1 + ((size_t)i * 64 + tid) * (D_MODEL + 2);
            for (int k = 0; k < D_MODEL; ++k) a += fr[i][k] * w[k];
            a += lg[i][0] * w[D_MODEL] + lg[i][1] * w[D_MODEL + 1];
            h1[tid] = fmaxf(a, 0.0f);
        }
        __syncthreads();
        if (tid == 0) {
            float a = conf_b2[i];
            const float* w2 = conf_w2 + (size_t)i * 64;
            for (int k = 0; k < 64; ++k) a += h1[k] * w2[k];
            cf[i] = 1.0f / (1.0f + expf(-a));
        }
        __syncthreads();
    }

    if (tid == 0) {
        bool e0 = cf[0] >= 0.85f;
        bool e1 = cf[1] >= 0.85f;
        int ei = e0 ? 0 : (e1 ? 1 : 2);
        out[(size_t)b * 2 + 0] = lg[ei][0];
        out[(size_t)b * 2 + 1] = lg[ei][1];
        out[(size_t)BATCH * 2 + b] = (float)ei;   // exit_idx
    }
}

// =================================================================
// host-side launch
// =================================================================
static inline void launch_gemm(const float* A, int lda, const _Float16* Wh,
                               const float* bias, float* C, int ldc,
                               int M, int N, int K, int act, hipStream_t stream) {
    dim3 grid((N + NT - 1) / NT, (M + MT - 1) / MT);
    wmma_gemm_kernel<<<grid, 256, 0, stream>>>(A, lda, Wh, bias, C, ldc, M, N, K, act);
}

static inline void launch_cvt(const float* src, int Ks, _Float16* dst, int Kd,
                              int rows, hipStream_t stream) {
    int total = rows * Kd;
    cvt_pad_kernel<<<(total + 255) / 256, 256, 0, stream>>>(src, Ks, dst, Kd, rows);
}

extern "C" void kernel_launch(void* const* d_in, const int* in_sizes, int n_in,
                              void* d_out, int out_size, void* d_ws, size_t ws_size,
                              hipStream_t stream) {
    (void)in_sizes; (void)n_in; (void)out_size; (void)ws_size;
    const float* x          = (const float*)d_in[0];
    const float* emb_proto  = (const float*)d_in[1];
    const float* emb_flags  = (const float*)d_in[2];
    const float* emb_dir    = (const float*)d_in[3];
    const float* proj_len_w = (const float*)d_in[4];
    const float* proj_len_b = (const float*)d_in[5];
    const float* proj_iat_w = (const float*)d_in[6];
    const float* proj_iat_b = (const float*)d_in[7];
    const float* fusion_w   = (const float*)d_in[8];
    const float* fusion_b   = (const float*)d_in[9];
    const float* tok_ln_g   = (const float*)d_in[10];
    const float* tok_ln_b   = (const float*)d_in[11];
    const float* in_proj_w  = (const float*)d_in[12];
    const float* conv_w     = (const float*)d_in[13];
    const float* conv_b     = (const float*)d_in[14];
    const float* x_proj_w   = (const float*)d_in[15];
    const float* dt_proj_w  = (const float*)d_in[16];
    const float* dt_proj_b  = (const float*)d_in[17];
    const float* A_log      = (const float*)d_in[18];
    const float* D_param    = (const float*)d_in[19];
    const float* out_proj_w = (const float*)d_in[20];
    const float* ln_g       = (const float*)d_in[21];
    const float* ln_b       = (const float*)d_in[22];
    const float* clf_w1     = (const float*)d_in[23];
    const float* clf_b1     = (const float*)d_in[24];
    const float* clf_w2     = (const float*)d_in[25];
    const float* clf_b2     = (const float*)d_in[26];
    const float* conf_w1    = (const float*)d_in[27];
    const float* conf_b1    = (const float*)d_in[28];
    const float* conf_w2    = (const float*)d_in[29];
    const float* conf_b2    = (const float*)d_in[30];

    // ---- workspace carve-up ----
    float* ws = (float*)d_ws;
    size_t off = 0;
    float* feat = ws + off; off += (size_t)NTOK * D_MODEL;        // 64 MB
    float* xz   = ws + off; off += (size_t)NTOK * 2 * D_INNER;    // 256 MB
    float* dbl  = ws + off; off += (size_t)NTOK * (DT_RANK + 2 * D_STATE);
    float* dt16 = ws + off; off += (size_t)NTOK * 32;             // padded dt input
    float* dty  = ws + off; off += (size_t)NTOK * D_INNER;        // dt, then y in place
    float* mm   = ws + off; off += (size_t)NTOK * D_MODEL;
    float* cat  = xz;  // cat (65536x160) aliases xz: used only before layer 0

    // f16 weight area (16B aligned: off is a multiple of NTOK)
    _Float16* wh = (_Float16*)(ws + off);
    size_t h = 0;
    _Float16* fusion_h  = wh + h; h += (size_t)D_MODEL * D_CATP;                      // 256x160
    _Float16* in_proj_h = wh + h; h += (size_t)N_LAYERS * 2 * D_INNER * D_MODEL;      // 4x1024x256
    _Float16* x_proj_h  = wh + h; h += (size_t)N_LAYERS * (DT_RANK + 2*D_STATE) * D_INNER;
    _Float16* dt_proj_h = wh + h; h += (size_t)N_LAYERS * D_INNER * 32;               // padded K=32
    _Float16* out_proj_h= wh + h; h += (size_t)N_LAYERS * D_MODEL * D_INNER;

    // 0) weight conversion to f16 (with K padding where needed)
    launch_cvt(fusion_w,  D_CAT,   fusion_h,  D_CATP,  D_MODEL, stream);
    launch_cvt(in_proj_w, D_MODEL, in_proj_h, D_MODEL, N_LAYERS * 2 * D_INNER, stream);
    launch_cvt(x_proj_w,  D_INNER, x_proj_h,  D_INNER, N_LAYERS * (DT_RANK + 2*D_STATE), stream);
    launch_cvt(dt_proj_w, DT_RANK, dt_proj_h, 32,      N_LAYERS * D_INNER, stream);
    launch_cvt(out_proj_w,D_INNER, out_proj_h,D_INNER, N_LAYERS * D_MODEL, stream);

    // 1) fused input features (padded to 160)
    {
        int total = NTOK * D_CATP;
        build_cat_kernel<<<(total + 255) / 256, 256, 0, stream>>>(
            x, emb_proto, emb_flags, emb_dir,
            proj_len_w, proj_len_b, proj_iat_w, proj_iat_b, cat);
    }
    // 2) fusion GEMM + token LN
    launch_gemm(cat, D_CATP, fusion_h, fusion_b, mm, D_MODEL,
                NTOK, D_MODEL, D_CATP, 0, stream);
    add_ln_kernel<<<NTOK / 8, 256, 0, stream>>>(mm, nullptr, tok_ln_g, tok_ln_b, feat);

    // 3) Mamba layers
    for (int l = 0; l < N_LAYERS; ++l) {
        const _Float16* ipw = in_proj_h  + (size_t)l * 2 * D_INNER * D_MODEL;
        const float*    cw  = conv_w     + (size_t)l * D_INNER * D_CONV;
        const float*    cb  = conv_b     + (size_t)l * D_INNER;
        const _Float16* xpw = x_proj_h   + (size_t)l * (DT_RANK + 2 * D_STATE) * D_INNER;
        const _Float16* dpw = dt_proj_h  + (size_t)l * D_INNER * 32;
        const float*    dpb = dt_proj_b  + (size_t)l * D_INNER;
        const float*    Al  = A_log      + (size_t)l * D_INNER * D_STATE;
        const float*    Dpl = D_param    + (size_t)l * D_INNER;
        const _Float16* opw = out_proj_h + (size_t)l * D_MODEL * D_INNER;

        // xz = feat @ in_proj^T    (65536 x 1024)
        launch_gemm(feat, D_MODEL, ipw, nullptr, xz, 2 * D_INNER,
                    NTOK, 2 * D_INNER, D_MODEL, 0, stream);
        // depthwise causal conv + SiLU on xi (in place)
        conv_silu_kernel<<<(BATCH * D_INNER + 255) / 256, 256, 0, stream>>>(xz, cw, cb);
        // dbl = xi @ x_proj^T      (65536 x 48), xi strided inside xz
        launch_gemm(xz, 2 * D_INNER, xpw, nullptr, dbl, DT_RANK + 2 * D_STATE,
                    NTOK, DT_RANK + 2 * D_STATE, D_INNER, 0, stream);
        // repack dt-rank slice into padded K=32 buffer
        repack_dt_kernel<<<(NTOK * 32 + 255) / 256, 256, 0, stream>>>(dbl, dt16);
        // dt = softplus(dt16 @ dt_proj^T + dpb)   (65536 x 512)
        launch_gemm(dt16, 32, dpw, dpb, dty, D_INNER,
                    NTOK, D_INNER, 32, 1, stream);
        // selective scan + D skip + SiLU gate -> y (in place over dt)
        scan_kernel<<<(BATCH * D_INNER + 255) / 256, 256, 0, stream>>>(xz, dbl, dty, Al, Dpl);
        // mm = y @ out_proj^T      (65536 x 256)
        launch_gemm(dty, D_INNER, opw, nullptr, mm, D_MODEL,
                    NTOK, D_MODEL, D_INNER, 0, stream);
        // feat = LN(mm + feat)
        add_ln_kernel<<<NTOK / 8, 256, 0, stream>>>(mm, feat, ln_g, ln_b, feat);
    }

    // 4) early-exit heads
    heads_kernel<<<BATCH, 128, 0, stream>>>(feat, clf_w1, clf_b1, clf_w2, clf_b2,
                                            conf_w1, conf_b1, conf_w2, conf_b2,
                                            (float*)d_out);
}
